// TransFusionHead_54125177864743
// MI455X (gfx1250) — compile-verified, matrix operand order checked by reading, not run on hardware
//
#include <hip/hip_runtime.h>
#include <hip/hip_bf16.h>
#include <stdint.h>

// ---------------- constants ----------------
#define BB    2
#define CIN   512
#define HH    180
#define WW    180
#define HWSZ  32400
#define HC    128
#define NCLS  10
#define NQ    200
#define NHEAD 8
#define HD    16
#define EPSV  1e-5f

// try the CDNA5 async global->LDS path (flip to 0 if the assembler rejects it)
#define USE_ASYNC_LDS 1

typedef __attribute__((ext_vector_type(16))) _Float16 v16h;
typedef __attribute__((ext_vector_type(8)))  float    v8f;

// ---------------- helpers ----------------
__device__ __forceinline__ float sigmoidf(float x) { return 1.0f / (1.0f + __expf(-x)); }

// ============================================================
// Weight prepack into WMMA A-fragment order (f16).
// layout: [(tap*KC + kc)*8 + mtile][lane(32)][elem(16)]
// lane l: M = l%16, g = l/16 ; elem e -> K = g*8+e (e<8) else 16+g*8+(e-8)
// ============================================================
__global__ void pack_conv_w(const float* __restrict__ w, _Float16* __restrict__ out,
                            int IC, int KC, int total) {
  int t = blockIdx.x * 256 + threadIdx.x;
  if (t >= total) return;
  int e    = t & 15;
  int lane = (t >> 4) & 31;
  int mt   = (t >> 9) & 7;
  int r    = t >> 12;          // tap*KC + kc
  int kc   = r % KC;
  int tap  = r / KC;
  int g = lane >> 4, m = lane & 15;
  int kk = (e < 8) ? (g * 8 + e) : (16 + g * 8 + (e - 8));
  int ic = kc * 32 + kk;
  int oc = mt * 16 + m;
  int ky = tap / 3, kx = tap % 3;
  out[t] = (_Float16)w[(((size_t)oc * IC + ic) * 3 + ky) * 3 + kx];
}

// plain f32 -> f16 pack
__global__ void pack_f16_k(const float* __restrict__ in, _Float16* __restrict__ out, int n) {
  int t = blockIdx.x * 256 + threadIdx.x;
  if (t < n) out[t] = (_Float16)in[t];
}

// ============================================================
// 3x3 SAME conv, OC=128, implicit-GEMM WMMA f16, row-staged LDS.
// EPI=0: +bias(p0). EPI=1: BN(g=p0,b=p1,m=p2,v=p3)+ReLU.
// grid = B*H*12 blocks of 256 threads (8 waves).
// LDS holds one full input row (IC x 18 f32); async DMA'd when possible.
// ============================================================
template <int IC, int EPI>
__global__ __launch_bounds__(256) void conv3x3_wmma(
    const float* __restrict__ xin, const _Float16* __restrict__ wp,
    const float* __restrict__ p0, const float* __restrict__ p1,
    const float* __restrict__ p2, const float* __restrict__ p3,
    float* __restrict__ out) {
  constexpr int KC = IC / 32;
  __shared__ float lsf[IC * 18];   // row cache: [ic][xx], xx = x0-1 .. x0+16

  int blk = blockIdx.x;
  int xt = blk % 12;
  int y  = (blk / 12) % HH;
  int b  = blk / (12 * HH);
  int x0 = xt * 16;

  int wave = threadIdx.x >> 5, lane = threadIdx.x & 31;
  int g = lane >> 4, n16 = lane & 15;

  v8f acc;
#pragma unroll
  for (int r = 0; r < 8; ++r) acc[r] = 0.0f;

  for (int ky = 0; ky < 3; ++ky) {
    int yy = y + ky - 1;
    bool rowvalid = (yy >= 0 && yy < HH);
    __syncthreads();                               // previous row fully consumed
    for (int idx = threadIdx.x; idx < IC * 18; idx += 256) {
      int ic = idx / 18, xx = idx % 18;
      int gx = x0 + xx - 1;
      bool inr = rowvalid && gx >= 0 && gx < WW;
#if USE_ASYNC_LDS
      if (inr) {
        const float* gp = &xin[(((size_t)b * IC + ic) * HH + yy) * WW + gx];
        unsigned int loff = (unsigned int)(uintptr_t)&lsf[idx];   // low 32 bits = LDS offset
        unsigned long long ga = (unsigned long long)(uintptr_t)gp;
        asm volatile("global_load_async_to_lds_b32 %0, %1, off"
                     :: "v"(loff), "v"(ga) : "memory");
      } else {
        lsf[idx] = 0.0f;
      }
#else
      float v = 0.0f;
      if (inr) v = xin[(((size_t)b * IC + ic) * HH + yy) * WW + gx];
      lsf[idx] = v;
#endif
    }
#if USE_ASYNC_LDS
    asm volatile("s_wait_asynccnt 0" ::: "memory");
#endif
    __syncthreads();

    for (int kx = 0; kx < 3; ++kx) {
      int tap = ky * 3 + kx;
      for (int kc = 0; kc < KC; ++kc) {
        const _Float16* ap = wp + (((size_t)(tap * KC + kc) * 8 + wave) * 512) + lane * 16;
        v16h av = *(const v16h*)ap;
        v16h bv;
#pragma unroll
        for (int e = 0; e < 16; ++e) {
          int kk = (e < 8) ? (g * 8 + e) : (16 + g * 8 + (e - 8));
          bv[e] = (_Float16)lsf[(kc * 32 + kk) * 18 + n16 + kx];
        }
        acc = __builtin_amdgcn_wmma_f32_16x16x32_f16(false, av, false, bv,
                                                     (short)0, acc, false, false);
      }
    }
  }

  int half = lane >> 4, nn = lane & 15;
  int px = x0 + nn;
  if (px < WW) {
#pragma unroll
    for (int r = 0; r < 8; ++r) {
      int m  = r + half * 8;
      int oc = wave * 16 + m;
      float o;
      if (EPI == 0) {
        o = acc[r] + p0[oc];
      } else {
        float s = p0[oc] * rsqrtf(p3[oc] + EPSV);
        o = (acc[r] - p2[oc]) * s + p1[oc];
        o = fmaxf(o, 0.0f);
      }
      out[(size_t)(b * HC + oc) * HWSZ + y * WW + px] = o;
    }
  }
}

// ============================================================
// 3x3 conv 128 -> 10 (+bias), direct.
// ============================================================
__global__ void conv_hm2_k(const float* __restrict__ h, const float* __restrict__ w,
                           const float* __restrict__ bias, float* __restrict__ dense) {
  int t = blockIdx.x * 256 + threadIdx.x;
  if (t >= BB * NCLS * HWSZ) return;
  int p  = t % HWSZ;
  int oc = (t / HWSZ) % NCLS;
  int b  = t / (NCLS * HWSZ);
  int y = p / WW, x = p % WW;
  float s = bias[oc];
  for (int c = 0; c < HC; ++c) {
    const float* hp = h + (size_t)(b * HC + c) * HWSZ;
    const float* wp = w + ((size_t)oc * HC + c) * 9;
    for (int ky = 0; ky < 3; ++ky) {
      int yy = y + ky - 1;
      if (yy < 0 || yy >= HH) continue;
      for (int kx = 0; kx < 3; ++kx) {
        int xx = x + kx - 1;
        if (xx < 0 || xx >= WW) continue;
        s += wp[ky * 3 + kx] * hp[yy * WW + xx];
      }
    }
  }
  dense[t] = s;
}

// ============================================================
// sigmoid + 3x3 local-max NMS
// ============================================================
__global__ void nms_k(const float* __restrict__ dense, float* __restrict__ heat,
                      float* __restrict__ hscr) {
  int t = blockIdx.x * 256 + threadIdx.x;
  if (t >= BB * NCLS * HWSZ) return;
  int p = t % HWSZ;
  int y = p / WW, x = p % WW;
  const float* d0 = dense + (t - p);
  float s = sigmoidf(d0[p]);
  bool ok = true;
  for (int ky = -1; ky <= 1; ++ky) {
    int yy = y + ky;
    if (yy < 0 || yy >= HH) continue;
    for (int kx = -1; kx <= 1; ++kx) {
      int xx = x + kx;
      if (xx < 0 || xx >= WW) continue;
      float sn = sigmoidf(d0[yy * WW + xx]);
      if (sn > s) ok = false;
    }
  }
  float r = ok ? s : 0.0f;
  heat[t] = r;
  hscr[t] = r;
}

// ============================================================
// top-200 over class*HW per batch (iterative argmax, ties -> lower index)
// ============================================================
__global__ __launch_bounds__(256) void topk_k(float* __restrict__ hscr,
                                              int* __restrict__ idx, int* __restrict__ cls,
                                              float* __restrict__ qs) {
  int b = blockIdx.x;
  float* base = hscr + (size_t)b * NCLS * HWSZ;
  __shared__ float sv[256];
  __shared__ int si[256];
  for (int it = 0; it < NQ; ++it) {
    float bv = -1e30f;
    int bi = 0x7fffffff;
    for (int i = threadIdx.x; i < NCLS * HWSZ; i += 256) {
      float v = base[i];
      if (v > bv || (v == bv && i < bi)) { bv = v; bi = i; }
    }
    sv[threadIdx.x] = bv; si[threadIdx.x] = bi;
    __syncthreads();
    for (int off = 128; off > 0; off >>= 1) {
      if (threadIdx.x < off) {
        float v2 = sv[threadIdx.x + off];
        int   i2 = si[threadIdx.x + off];
        if (v2 > sv[threadIdx.x] || (v2 == sv[threadIdx.x] && i2 < si[threadIdx.x])) {
          sv[threadIdx.x] = v2; si[threadIdx.x] = i2;
        }
      }
      __syncthreads();
    }
    if (threadIdx.x == 0) {
      int top = si[0];
      idx[b * NQ + it] = top % HWSZ;
      cls[b * NQ + it] = top / HWSZ;
      qs[b * NQ + it]  = sv[0];
      base[top] = -1e30f;
    }
    __syncthreads();
  }
}

// ============================================================
// gather query features + class embedding, and query positions
// ============================================================
__global__ __launch_bounds__(128) void gather_k(const float* __restrict__ ff,
                                                const int* __restrict__ idx, const int* __restrict__ cls,
                                                const float* __restrict__ cew, const float* __restrict__ ceb,
                                                float* __restrict__ tgt0, float* __restrict__ qp) {
  int bn = blockIdx.x;
  int b = bn / NQ, n = bn % NQ;
  int c = threadIdx.x;
  int pix = idx[bn];
  int cl  = cls[bn];
  tgt0[(size_t)bn * HC + c] = ff[(size_t)(b * HC + c) * HWSZ + pix] + cew[c * NCLS + cl] + ceb[c];
  if (c == 0) {
    qp[(b * 2 + 0) * NQ + n] = (float)(pix / WW) + 0.5f;
    qp[(b * 2 + 1) * NQ + n] = (float)(pix % WW) + 0.5f;
  }
}

// ============================================================
// posembed for queries (sp_*) -> qpe [B,NQ,128]
// ============================================================
__global__ __launch_bounds__(128) void posembed_q_k(
    const float* __restrict__ qp, const float* __restrict__ w1, const float* __restrict__ b1,
    const float* __restrict__ g, const float* __restrict__ bb, const float* __restrict__ m,
    const float* __restrict__ vv, const float* __restrict__ w2, const float* __restrict__ b2,
    float* __restrict__ out) {
  int bn = blockIdx.x;
  int b = bn / NQ, n = bn % NQ;
  int c = threadIdx.x;
  __shared__ float h[HC];
  float p0 = qp[(b * 2 + 0) * NQ + n], p1 = qp[(b * 2 + 1) * NQ + n];
  float t = w1[c * 2] * p0 + w1[c * 2 + 1] * p1 + b1[c];
  float s = g[c] * rsqrtf(vv[c] + EPSV);
  h[c] = fmaxf((t - m[c]) * s + bb[c], 0.0f);
  __syncthreads();
  float o = b2[c];
  for (int k = 0; k < HC; ++k) o += w2[c * HC + k] * h[k];
  out[(size_t)bn * HC + c] = o;
}

// ============================================================
// BEV posembed (cp_*), add to mem, emit f16 K-input and V-input rows
// ============================================================
__global__ __launch_bounds__(128) void kbuild_k(
    const float* __restrict__ ff, const float* __restrict__ w1, const float* __restrict__ b1,
    const float* __restrict__ g, const float* __restrict__ bb, const float* __restrict__ m,
    const float* __restrict__ vv, const float* __restrict__ w2, const float* __restrict__ b2,
    _Float16* __restrict__ kbf, _Float16* __restrict__ vbf) {
  int bp = blockIdx.x;
  int b = bp / HWSZ, p = bp % HWSZ;
  int c = threadIdx.x;
  __shared__ float h[HC];
  float p0 = (float)(p / WW) + 0.5f, p1 = (float)(p % WW) + 0.5f;
  float t = w1[c * 2] * p0 + w1[c * 2 + 1] * p1 + b1[c];
  float s = g[c] * rsqrtf(vv[c] + EPSV);
  h[c] = fmaxf((t - m[c]) * s + bb[c], 0.0f);
  __syncthreads();
  float kpe = b2[c];
  for (int k = 0; k < HC; ++k) kpe += w2[c * HC + k] * h[k];
  float mv = ff[(size_t)(b * HC + c) * HWSZ + p];
  kbf[(size_t)bp * HC + c] = (_Float16)(mv + kpe);
  vbf[(size_t)bp * HC + c] = (_Float16)mv;
}

// ============================================================
// wave-level WMMA GEMM:  C[m,n] = sum_k A[m,k]*W[n,k] + bias[n]  (K=N=128)
// A f16 row-major [Mtot,128]; W f16 [128,128]; out f16 [Mtot,128].
// grid = (Mtot/16)*8 blocks of 32 threads.
// ============================================================
__global__ __launch_bounds__(32) void gemm_wave_f16(
    const _Float16* __restrict__ A, const _Float16* __restrict__ Wt,
    const float* __restrict__ bias, _Float16* __restrict__ out) {
  int blk = blockIdx.x;
  int nt = blk & 7;
  int mt = blk >> 3;
  int lane = threadIdx.x & 31;
  int g = lane >> 4;
  int mrow = mt * 16 + (lane & 15);
  int ncol = nt * 16 + (lane & 15);
  v8f acc;
#pragma unroll
  for (int r = 0; r < 8; ++r) acc[r] = 0.0f;
  for (int kc = 0; kc < 4; ++kc) {
    v16h av, bv;
#pragma unroll
    for (int e = 0; e < 16; ++e) {
      int kk = (e < 8) ? (g * 8 + e) : (16 + g * 8 + (e - 8));
      av[e] = A[(size_t)mrow * HC + kc * 32 + kk];
      bv[e] = Wt[(size_t)ncol * HC + kc * 32 + kk];
    }
    acc = __builtin_amdgcn_wmma_f32_16x16x32_f16(false, av, false, bv,
                                                 (short)0, acc, false, false);
  }
  int half = lane >> 4, nn = lane & 15;
#pragma unroll
  for (int r = 0; r < 8; ++r) {
    int mm = mt * 16 + r + half * 8;
    out[(size_t)mm * HC + nt * 16 + nn] = (_Float16)(acc[r] + bias[nt * 16 + nn]);
  }
}

// ============================================================
// self-attn projections: q (scaled 1/4), k from tgt+qpe ; v from tgt
// ============================================================
__global__ __launch_bounds__(128) void saproj_k(
    const float* __restrict__ tgt0, const float* __restrict__ qpe,
    const float* __restrict__ Wt, const float* __restrict__ bvec,
    float* __restrict__ sq, float* __restrict__ sk, float* __restrict__ sv) {
  int bn = blockIdx.x;
  int j = threadIdx.x;
  __shared__ float t[HC], tq[HC];
  float tv = tgt0[(size_t)bn * HC + j];
  t[j] = tv;
  tq[j] = tv + qpe[(size_t)bn * HC + j];
  __syncthreads();
  float aq = bvec[j], ak = bvec[HC + j], av = bvec[2 * HC + j];
  for (int c = 0; c < HC; ++c) {
    aq += Wt[(size_t)j * HC + c] * tq[c];
    ak += Wt[(size_t)(HC + j) * HC + c] * tq[c];
    av += Wt[(size_t)(2 * HC + j) * HC + c] * t[c];
  }
  sq[(size_t)bn * HC + j] = aq * 0.25f;
  sk[(size_t)bn * HC + j] = ak;
  sv[(size_t)bn * HC + j] = av;
}

// ============================================================
// self-attention, one thread per (b,h,q), 200 keys
// ============================================================
__global__ void selfattn_k(const float* __restrict__ sq, const float* __restrict__ sk,
                           const float* __restrict__ sv, float* __restrict__ so) {
  int t = blockIdx.x * blockDim.x + threadIdx.x;
  if (t >= BB * NHEAD * NQ) return;
  int b = t / (NHEAD * NQ);
  int h = (t / NQ) % NHEAD;
  int n = t % NQ;
  float q[HD];
#pragma unroll
  for (int j = 0; j < HD; ++j) q[j] = sq[(size_t)(b * NQ + n) * HC + h * HD + j];
  float mx = -1e30f;
  for (int k = 0; k < NQ; ++k) {
    const float* kp = sk + (size_t)(b * NQ + k) * HC + h * HD;
    float s = 0.0f;
#pragma unroll
    for (int j = 0; j < HD; ++j) s += q[j] * kp[j];
    mx = fmaxf(mx, s);
  }
  float l = 0.0f, acc[HD];
#pragma unroll
  for (int j = 0; j < HD; ++j) acc[j] = 0.0f;
  for (int k = 0; k < NQ; ++k) {
    const float* kp = sk + (size_t)(b * NQ + k) * HC + h * HD;
    float s = 0.0f;
#pragma unroll
    for (int j = 0; j < HD; ++j) s += q[j] * kp[j];
    float e = __expf(s - mx);
    l += e;
    const float* vp = sv + (size_t)(b * NQ + k) * HC + h * HD;
#pragma unroll
    for (int j = 0; j < HD; ++j) acc[j] += e * vp[j];
  }
#pragma unroll
  for (int j = 0; j < HD; ++j) so[(size_t)(b * NQ + n) * HC + h * HD + j] = acc[j] / l;
}

// ============================================================
// out-projection + residual + LayerNorm (K = 128 or 256)
// ============================================================
__global__ __launch_bounds__(128) void addnorm_k(
    const float* __restrict__ resid, const float* __restrict__ feat,
    const float* __restrict__ Wt, const float* __restrict__ pb,
    const float* __restrict__ g, const float* __restrict__ bb,
    float* __restrict__ out, int K) {
  __shared__ float sf[256];
  __shared__ float red[128];
  __shared__ float mu_s, var_s;
  int bn = blockIdx.x;
  int c = threadIdx.x;
  for (int i = c; i < K; i += 128) sf[i] = feat[(size_t)bn * K + i];
  __syncthreads();
  float d = resid[(size_t)bn * HC + c] + pb[c];
  for (int k = 0; k < K; ++k) d += Wt[(size_t)c * K + k] * sf[k];
  red[c] = d;
  __syncthreads();
  for (int off = 64; off > 0; off >>= 1) {
    if (c < off) red[c] += red[c + off];
    __syncthreads();
  }
  if (c == 0) mu_s = red[0] * (1.0f / 128.0f);
  __syncthreads();
  float dm = d - mu_s;
  red[c] = dm * dm;
  __syncthreads();
  for (int off = 64; off > 0; off >>= 1) {
    if (c < off) red[c] += red[c + off];
    __syncthreads();
  }
  if (c == 0) var_s = red[0] * (1.0f / 128.0f);
  __syncthreads();
  out[(size_t)bn * HC + c] = dm * rsqrtf(var_s + EPSV) * g[c] + bb[c];
}

// ============================================================
// cross-attn query projection (scaled 1/4)
// ============================================================
__global__ __launch_bounds__(128) void caqproj_k(
    const float* __restrict__ tgt1, const float* __restrict__ qpe,
    const float* __restrict__ Wt, const float* __restrict__ bvec, float* __restrict__ qca) {
  int bn = blockIdx.x;
  int j = threadIdx.x;
  __shared__ float tq[HC];
  tq[j] = tgt1[(size_t)bn * HC + j] + qpe[(size_t)bn * HC + j];
  __syncthreads();
  float s = bvec[j];
  for (int c = 0; c < HC; ++c) s += Wt[(size_t)j * HC + c] * tq[c];
  qca[(size_t)bn * HC + j] = s * 0.25f;
}

// ============================================================
// cross-attention, one wave per (b,h,q), lanes stripe 32400 keys (online softmax)
// ============================================================
__global__ __launch_bounds__(32) void crossattn_k(
    const float* __restrict__ qca, const _Float16* __restrict__ kproj,
    const _Float16* __restrict__ vproj, float* __restrict__ cao) {
  int blk = blockIdx.x;
  int b = blk / (NHEAD * NQ);
  int h = (blk / NQ) % NHEAD;
  int n = blk % NQ;
  int lane = threadIdx.x;
  float q[HD];
#pragma unroll
  for (int j = 0; j < HD; ++j) q[j] = qca[(size_t)(b * NQ + n) * HC + h * HD + j];
  float m = -1e30f, l = 0.0f, acc[HD];
#pragma unroll
  for (int j = 0; j < HD; ++j) acc[j] = 0.0f;
  for (int p = lane; p < HWSZ; p += 32) {
    const _Float16* kp = kproj + (size_t)(b * HWSZ + p) * HC + h * HD;
    const _Float16* vp = vproj + (size_t)(b * HWSZ + p) * HC + h * HD;
    __builtin_prefetch(kp + 32 * HC, 0, 1);
    float s = 0.0f;
#pragma unroll
    for (int j = 0; j < HD; ++j) s += q[j] * (float)kp[j];
    float mn = fmaxf(m, s);
    float co = __expf(m - mn);
    float e  = __expf(s - mn);
    l = l * co + e;
#pragma unroll
    for (int j = 0; j < HD; ++j) acc[j] = acc[j] * co + e * (float)vp[j];
    m = mn;
  }
  for (int off = 16; off > 0; off >>= 1) {
    float m2 = __shfl_down(m, off, 32);
    float l2 = __shfl_down(l, off, 32);
    float mn = fmaxf(m, m2);
    float a = __expf(m - mn), b2 = __expf(m2 - mn);
#pragma unroll
    for (int j = 0; j < HD; ++j) {
      float a2 = __shfl_down(acc[j], off, 32);
      acc[j] = acc[j] * a + a2 * b2;
    }
    l = l * a + l2 * b2;
    m = mn;
  }
  if (lane == 0) {
#pragma unroll
    for (int j = 0; j < HD; ++j)
      cao[(size_t)(b * NQ + n) * HC + h * HD + j] = acc[j] / l;
  }
}

// ============================================================
// FFN first layer + ReLU (128 -> 256)
// ============================================================
__global__ __launch_bounds__(256) void ffn1_k(const float* __restrict__ tgt2,
                                              const float* __restrict__ w1,
                                              const float* __restrict__ b1,
                                              float* __restrict__ hid) {
  int bn = blockIdx.x;
  int j = threadIdx.x;
  __shared__ float t[HC];
  if (j < HC) t[j] = tgt2[(size_t)bn * HC + j];
  __syncthreads();
  float s = b1[j];
  for (int c = 0; c < HC; ++c) s += w1[(size_t)j * HC + c] * t[c];
  hid[(size_t)bn * 256 + j] = fmaxf(s, 0.0f);
}

// ============================================================
// prediction head: conv1d->BN->ReLU->conv1d (+ optional qp add for center)
// ============================================================
__global__ __launch_bounds__(64) void head_k(
    const float* __restrict__ tgt, const float* __restrict__ w1, const float* __restrict__ g,
    const float* __restrict__ bb, const float* __restrict__ m, const float* __restrict__ vv,
    const float* __restrict__ w2, const float* __restrict__ b2, int oc,
    float* __restrict__ out, const float* __restrict__ qp) {
  int bn = blockIdx.x;
  int b = bn / NQ, n = bn % NQ;
  int o = threadIdx.x;
  __shared__ float t[HC];
  __shared__ float h1[64];
  t[o] = tgt[(size_t)bn * HC + o];
  t[o + 64] = tgt[(size_t)bn * HC + o + 64];
  __syncthreads();
  float s = 0.0f;
  for (int c = 0; c < HC; ++c) s += w1[(size_t)o * HC + c] * t[c];
  float sc = g[o] * rsqrtf(vv[o] + EPSV);
  h1[o] = fmaxf((s - m[o]) * sc + bb[o], 0.0f);
  __syncthreads();
  if (o < oc) {
    float r = b2[o];
    for (int k = 0; k < 64; ++k) r += w2[(size_t)o * 64 + k] * h1[k];
    if (qp != nullptr) r += qp[(b * 2 + o) * NQ + n];
    out[(size_t)(b * oc + o) * NQ + n] = r;
  }
}

// ============================================================
// cls (as float) + qs [B,NC,NQ] gather
// ============================================================
__global__ void finalout_k(const int* __restrict__ cls, const int* __restrict__ idx,
                           const float* __restrict__ heat, float* __restrict__ out_cls,
                           float* __restrict__ out_qs) {
  int t = blockIdx.x * 256 + threadIdx.x;
  if (t < BB * NQ) out_cls[t] = (float)cls[t];
  if (t < BB * NCLS * NQ) {
    int b = t / (NCLS * NQ);
    int c = (t / NQ) % NCLS;
    int n = t % NQ;
    out_qs[t] = heat[(size_t)(b * NCLS + c) * HWSZ + idx[b * NQ + n]];
  }
}

// ============================================================
// host side
// ============================================================
extern "C" void kernel_launch(void* const* d_in, const int* in_sizes, int n_in,
                              void* d_out, int out_size, void* d_ws, size_t ws_size,
                              hipStream_t stream) {
  (void)in_sizes; (void)n_in; (void)out_size; (void)ws_size;
#define P(i) ((const float*)d_in[i])
  const float* x = P(0);
  // params in setup_inputs insertion order:
  // 1 sc_w 2 sc_b 3 hm1_w 4-7 hm_bn 8 hm2_w 9 hm2_b 10 ce_w 11 ce_b
  // 12 sa_in_w 13 sa_in_b 14 sa_out_w 15 sa_out_b 16 ca_in_w 17 ca_in_b
  // 18 ca_out_w 19 ca_out_b 20 ffn_w1 21 ffn_b1 22 ffn_w2 23 ffn_b2
  // 24-31 sp_* 32-39 cp_* 40-45 ln0..2 46.. heads (7 each)

  float* Wc = (float*)d_ws;
  float* ff    = Wc; Wc += (size_t)BB * HC * HWSZ;           // 8,294,400
  float* hbuf  = Wc; Wc += (size_t)BB * HC * HWSZ;           // reused as kbf/vbf later
  float* dense = Wc; Wc += (size_t)BB * NCLS * HWSZ;
  float* heat  = Wc; Wc += (size_t)BB * NCLS * HWSZ;
  float* hscr  = Wc; Wc += (size_t)BB * NCLS * HWSZ;
  _Float16* kproj = (_Float16*)Wc; Wc += (size_t)BB * HWSZ * HC / 2;
  _Float16* vproj = (_Float16*)Wc; Wc += (size_t)BB * HWSZ * HC / 2;
  _Float16* wpsc  = (_Float16*)Wc; Wc += 294912;             // 589,824 halves
  _Float16* wphm  = (_Float16*)Wc; Wc += 73728;              // 147,456 halves
  _Float16* wkv   = (_Float16*)Wc; Wc += 16384;              // 32,768 halves (K+V proj weights)
  int*   t_idx = (int*)Wc;  Wc += BB * NQ;
  int*   t_cls = (int*)Wc;  Wc += BB * NQ;
  float* t_qs  = Wc; Wc += BB * NQ;
  float* qp    = Wc; Wc += BB * 2 * NQ;
  float* tgt0  = Wc; Wc += BB * NQ * HC;
  float* qpe   = Wc; Wc += BB * NQ * HC;
  float* saq   = Wc; Wc += BB * NQ * HC;
  float* sak   = Wc; Wc += BB * NQ * HC;
  float* sav   = Wc; Wc += BB * NQ * HC;
  float* sao   = Wc; Wc += BB * NQ * HC;
  float* tgt1  = Wc; Wc += BB * NQ * HC;
  float* qca   = Wc; Wc += BB * NQ * HC;
  float* cao   = Wc; Wc += BB * NQ * HC;
  float* tgt2  = Wc; Wc += BB * NQ * HC;
  float* hid   = Wc; Wc += BB * NQ * 256;

  _Float16* kbf = (_Float16*)hbuf;                            // reuse hbuf region
  _Float16* vbf = kbf + (size_t)BB * HWSZ * HC;

  float* tgt_out = (float*)d_out;                 // 51200
  float* o_center = tgt_out + 51200;              // 800
  float* o_height = o_center + 800;               // 400
  float* o_dim    = o_height + 400;               // 1200
  float* o_rot    = o_dim + 1200;                 // 800
  float* o_vel    = o_rot + 800;                  // 800
  float* o_hm     = o_vel + 800;                  // 4000
  float* o_cls    = o_hm + 4000;                  // 400
  float* o_qs     = o_cls + 400;                  // 4000

  // 1) weight prepacks
  {
    int tot = 9 * 16 * 8 * 32 * 16;
    pack_conv_w<<<(tot + 255) / 256, 256, 0, stream>>>((const float*)d_in[1], wpsc, CIN, 16, tot);
    tot = 9 * 4 * 8 * 32 * 16;
    pack_conv_w<<<(tot + 255) / 256, 256, 0, stream>>>((const float*)d_in[3], wphm, HC, 4, tot);
    pack_f16_k<<<(32768 + 255) / 256, 256, 0, stream>>>(P(16) + (size_t)HC * HC, wkv, 32768);
  }

  // 2) fusion = conv(x) + bias   (WMMA)
  conv3x3_wmma<CIN, 0><<<BB * HH * 12, 256, 0, stream>>>(x, wpsc, P(2), P(2), P(2), P(2), ff);
  // 3) h = relu(bn(conv(fusion)))  (WMMA)
  conv3x3_wmma<HC, 1><<<BB * HH * 12, 256, 0, stream>>>(ff, wphm, P(4), P(5), P(6), P(7), hbuf);
  // 4) dense = conv(h) + bias
  conv_hm2_k<<<(BB * NCLS * HWSZ + 255) / 256, 256, 0, stream>>>(hbuf, P(8), P(9), dense);
  // 5) sigmoid + NMS
  nms_k<<<(BB * NCLS * HWSZ + 255) / 256, 256, 0, stream>>>(dense, heat, hscr);
  // 6) top-200
  topk_k<<<BB, 256, 0, stream>>>(hscr, t_idx, t_cls, t_qs);
  // 7) gather features + class embed + qp
  gather_k<<<BB * NQ, 128, 0, stream>>>(ff, t_idx, t_cls, P(10), P(11), tgt0, qp);
  // 8) query pos-embed (sp)
  posembed_q_k<<<BB * NQ, 128, 0, stream>>>(qp, P(24), P(25), P(26), P(27), P(28), P(29), P(30), P(31), qpe);
  // 9) key pos-embed (cp) + mem -> f16 K/V inputs (overwrites hbuf)
  kbuild_k<<<BB * HWSZ, 128, 0, stream>>>(ff, P(32), P(33), P(34), P(35), P(36), P(37), P(38), P(39), kbf, vbf);
  // 10) K/V projections (WMMA GEMM, f16 weights)
  {
    int mblocks = (BB * HWSZ / 16) * 8;
    gemm_wave_f16<<<mblocks, 32, 0, stream>>>(kbf, wkv, P(17) + HC, kproj);
    gemm_wave_f16<<<mblocks, 32, 0, stream>>>(vbf, wkv + (size_t)HC * HC, P(17) + 2 * HC, vproj);
  }
  // 11) self-attn
  saproj_k<<<BB * NQ, 128, 0, stream>>>(tgt0, qpe, P(12), P(13), saq, sak, sav);
  selfattn_k<<<(BB * NHEAD * NQ + 63) / 64, 64, 0, stream>>>(saq, sak, sav, sao);
  addnorm_k<<<BB * NQ, 128, 0, stream>>>(tgt0, sao, P(14), P(15), P(40), P(41), tgt1, HC);
  // 12) cross-attn
  caqproj_k<<<BB * NQ, 128, 0, stream>>>(tgt1, qpe, P(16), P(17), qca);
  crossattn_k<<<BB * NHEAD * NQ, 32, 0, stream>>>(qca, kproj, vproj, cao);
  addnorm_k<<<BB * NQ, 128, 0, stream>>>(tgt1, cao, P(18), P(19), P(42), P(43), tgt2, HC);
  // 13) FFN
  ffn1_k<<<BB * NQ, 256, 0, stream>>>(tgt2, P(20), P(21), hid);
  addnorm_k<<<BB * NQ, 128, 0, stream>>>(tgt2, hid, P(22), P(23), P(44), P(45), tgt_out, 256);
  // 14) heads (param blocks of 7 starting at 46)
  {
    const int ocs[6]   = {2, 1, 3, 2, 2, 10};
    float* outs[6]     = {o_center, o_height, o_dim, o_rot, o_vel, o_hm};
    for (int hix = 0; hix < 6; ++hix) {
      int base = 46 + hix * 7;
      head_k<<<BB * NQ, 64, 0, stream>>>(tgt_out, P(base), P(base + 1), P(base + 2),
                                         P(base + 3), P(base + 4), P(base + 5), P(base + 6),
                                         ocs[hix], outs[hix], hix == 0 ? qp : nullptr);
    }
  }
  // 15) cls + qs
  finalout_k<<<(BB * NCLS * NQ + 255) / 256, 256, 0, stream>>>(t_cls, t_idx, heat, o_cls, o_qs);
#undef P
}
